// DynamicTimeSeriesSelfAttention_42898133353110
// MI455X (gfx1250) — matrix operand order repaired and checked by reading, not compiled, not run on hardware
//
#include <hip/hip_runtime.h>

#define B_   4
#define H_   8
#define C_   64
#define L_   2048
#define BH_  (B_ * H_)     // 32
#define OC_  (C_ * H_)     // 512

typedef __attribute__((ext_vector_type(16))) _Float16 v16h;
typedef __attribute__((ext_vector_type(8)))  _Float16 v8h;
typedef __attribute__((ext_vector_type(8)))  float    v8f;

// Async global->LDS path (gfx1250 GLOBAL_LOAD_ASYNC_TO_LDS_B128, ASYNCcnt).
// Probe-derived signature: (int4-AS1*, int4-AS3*, imm, imm); int4 is the
// GCC vector_size flavor (per the round-2 diagnostic).
#if defined(__AMDGCN__) && __has_builtin(__builtin_amdgcn_global_load_async_to_lds_b128)
#define HAVE_ASYNC_LDS 1
typedef int v4i_gcc __attribute__((vector_size(16)));
typedef __attribute__((address_space(1))) v4i_gcc glb_v4i;
typedef __attribute__((address_space(3))) v4i_gcc lds_v4i;
#else
#define HAVE_ASYNC_LDS 0
#endif

__device__ __forceinline__ void async_wait_zero() {
#if __has_builtin(__builtin_amdgcn_s_wait_asynccnt)
  __builtin_amdgcn_s_wait_asynccnt(0);
#else
  asm volatile("s_wait_asynccnt 0x0" ::: "memory");
#endif
}

// Build a 16-half WMMA fragment register set from two contiguous 16B runs.
__device__ __forceinline__ v16h ld16(const _Float16* p0, const _Float16* p1) {
  v8h a = *(const v8h*)p0;
  v8h b = *(const v8h*)p1;
  v16h r;
#pragma unroll
  for (int i = 0; i < 8; ++i) { r[i] = a[i]; r[i + 8] = b[i]; }
  return r;
}

// ---------------------------------------------------------------------------
// QKV projection: out[oc,l] = sum_ic pw[oc,ic]*(x[ic,l]*dw[ic]+db[ic]) + pb[oc]
// mode 0/1 (Q,K): write f16 in (bh, l, c) layout (fragment-friendly)
// mode 2   (V)  : write f16 + f32 in (bh, c, l) layout
// ---------------------------------------------------------------------------
__global__ __launch_bounds__(256) void qkv_proj_kernel(
    const float* __restrict__ x, const float* __restrict__ dw,
    const float* __restrict__ db, const float* __restrict__ pw,
    const float* __restrict__ pb, _Float16* __restrict__ outH,
    float* __restrict__ outF, int mode) {
  int idx = blockIdx.x * blockDim.x + threadIdx.x;   // B_*OC_*L_ = 4194304
  int l  = idx & (L_ - 1);
  int oc = (idx >> 11) & (OC_ - 1);
  int b  = idx >> 20;

  float acc = pb[oc];
  const float* xr = x + (size_t)b * C_ * L_ + l;
  const float* wr = pw + oc * C_;
#pragma unroll 8
  for (int ic = 0; ic < C_; ++ic)
    acc = fmaf(wr[ic], fmaf(xr[(size_t)ic * L_], dw[ic], db[ic]), acc);

  int head = oc >> 6, c = oc & 63;
  int bh = b * H_ + head;
  if (mode < 2) {
    outH[((size_t)bh * L_ + l) * C_ + c] = (_Float16)acc;
  } else {
    size_t o = ((size_t)bh * C_ + c) * L_ + l;
    outH[o] = (_Float16)acc;
    outF[o] = acc;
  }
}

// ---------------------------------------------------------------------------
// Flash attention. Block = 8 waves, one (bh); wave w handles q-tile
// blockIdx.x*8+w. Each 32-key K/V tile is staged ONCE per block in LDS
// (async global->LDS when available) and shared by all 8 waves, cutting
// L2 read traffic 8x vs per-wave fragment loads.
// S[k,q] = (1/8) * sum_c K[c,k] Q[c,q] ; O[c,q] = sum_k V[c,k] softmax(S)[k,q]
// Qt/Kt stored (bh, l, c) f16 ; Vh stored (bh, c, l) f16 ; Ao (bh, c, l) f32.
// ---------------------------------------------------------------------------
__global__ __launch_bounds__(256) void flash_attn_kernel(
    const _Float16* __restrict__ Qt, const _Float16* __restrict__ Kt,
    const _Float16* __restrict__ Vh, float* __restrict__ Ao) {
  __shared__ _Float16 sK[32 * C_];   // [k within tile][c]      4 KB
  __shared__ _Float16 sV[C_ * 32];   // [c][k within tile]      4 KB

  const int tid   = threadIdx.x;
  const int lane  = tid & 31;
  const int wave  = tid >> 5;
  const int bh    = blockIdx.y;
  const int qtile = blockIdx.x * 8 + wave;
  const int qbase = qtile * 16;
  const int n = lane & 15;          // q column / B/D column
  const int g = lane >> 4;          // lane half

  // Q B-fragments: element e of lane -> Q[c = c0 + 8*g + e (e<8), c0 + 16 + 8*g + (e-8) (e>=8), q = qbase + n]
  const _Float16* Qrow = Qt + ((size_t)bh * L_ + qbase + n) * C_;
  const v16h qf0 = ld16(Qrow + 8 * g,      Qrow + 16 + 8 * g);
  const v16h qf1 = ld16(Qrow + 32 + 8 * g, Qrow + 48 + 8 * g);

  v8f o[4];
#pragma unroll
  for (int ct = 0; ct < 4; ++ct) o[ct] = (v8f){};
  float mrun = -3.0e38f, lrun = 0.0f;

  const _Float16* Kbase = Kt + (size_t)bh * L_ * C_;
  const _Float16* Vbase = Vh + (size_t)bh * C_ * L_;
  const float scale = 0.125f;   // c^-0.5, both c^-0.25 factors folded

  // Cooperative staging assignment (256 threads x 16B each per tile)
  const int kRow = tid >> 3, kCh = (tid & 7) * 8;     // K: 32 rows x 128B
  const int vRow = tid >> 2, vCh = (tid & 3) * 8;     // V: 64 rows x 64B

  for (int k0 = 0; k0 < L_; k0 += 32) {
    // ---- stage K/V tile into LDS (once per block) ----
    const _Float16* gK = Kbase + (size_t)(k0 + kRow) * C_ + kCh;
    const _Float16* gV = Vbase + (size_t)vRow * L_ + k0 + vCh;
    _Float16* lK = &sK[kRow * C_ + kCh];
    _Float16* lV = &sV[vRow * 32 + vCh];
#if HAVE_ASYNC_LDS
    __builtin_amdgcn_global_load_async_to_lds_b128((glb_v4i*)gK, (lds_v4i*)lK, 0, 0);
    __builtin_amdgcn_global_load_async_to_lds_b128((glb_v4i*)gV, (lds_v4i*)lV, 0, 0);
    async_wait_zero();
#else
    *(v8h*)lK = *(const v8h*)gK;
    *(v8h*)lV = *(const v8h*)gV;
#endif
    __syncthreads();

    // ---- S = K^T Q : A fragments from LDS K tile ----
    const _Float16* Kr0 = &sK[n * C_];
    const _Float16* Kr1 = &sK[(n + 16) * C_];
    v16h ka00 = ld16(Kr0 + 8 * g,      Kr0 + 16 + 8 * g);
    v16h ka01 = ld16(Kr0 + 32 + 8 * g, Kr0 + 48 + 8 * g);
    v16h ka10 = ld16(Kr1 + 8 * g,      Kr1 + 16 + 8 * g);
    v16h ka11 = ld16(Kr1 + 32 + 8 * g, Kr1 + 48 + 8 * g);

    v8f s0 = (v8f){}, s1 = (v8f){};
    s0 = __builtin_amdgcn_wmma_f32_16x16x32_f16(false, ka00, false, qf0, (short)0, s0, false, false);
    s0 = __builtin_amdgcn_wmma_f32_16x16x32_f16(false, ka01, false, qf1, (short)0, s0, false, false);
    s1 = __builtin_amdgcn_wmma_f32_16x16x32_f16(false, ka10, false, qf0, (short)0, s1, false, false);
    s1 = __builtin_amdgcn_wmma_f32_16x16x32_f16(false, ka11, false, qf1, (short)0, s1, false, false);

    // scale + tile max over k (both lane halves hold same q)
    float tmax = -3.0e38f;
#pragma unroll
    for (int r = 0; r < 8; ++r) {
      s0[r] *= scale; s1[r] *= scale;
      tmax = fmaxf(tmax, fmaxf(s0[r], s1[r]));
    }
    tmax = fmaxf(tmax, __shfl_xor(tmax, 16, 32));

    float mnew  = fmaxf(mrun, tmax);
    float alpha = __expf(mrun - mnew);
    lrun *= alpha;
#pragma unroll
    for (int ct = 0; ct < 4; ++ct)
#pragma unroll
      for (int r = 0; r < 8; ++r) o[ct][r] *= alpha;

    float p0[8], p1[8], psum = 0.0f;
#pragma unroll
    for (int r = 0; r < 8; ++r) {
      p0[r] = __expf(s0[r] - mnew);     // k = 8g + r        (within tile)
      p1[r] = __expf(s1[r] - mnew);     // k = 16 + 8g + r
      psum += p0[r] + p1[r];
    }
    psum += __shfl_xor(psum, 16, 32);
    lrun += psum;
    mrun  = mnew;

    // Repack P into B-fragment layout: element e of lane -> P[k = 16g + e, q = n]
    v16h pb;
#pragma unroll
    for (int r = 0; r < 8; ++r) {
      float q0 = __shfl_xor(p0[r], 16, 32);
      float q1 = __shfl_xor(p1[r], 16, 32);
      pb[r]     = (_Float16)(g ? q1 : p0[r]);   // low lanes k=r, high lanes k=16+r
      pb[r + 8] = (_Float16)(g ? p1[r] : q0);   // low lanes k=8+r, high lanes k=24+r
    }

    // O += V * P : A fragments of V from LDS (M = c tile, K = key tile of 32)
#pragma unroll
    for (int ct = 0; ct < 4; ++ct) {
      const _Float16* Vr = &sV[(ct * 16 + n) * 32];
      v16h va = ld16(Vr + 16 * g, Vr + 16 * g + 8);
      o[ct] = __builtin_amdgcn_wmma_f32_16x16x32_f16(false, va, false, pb, (short)0, o[ct], false, false);
    }
    __syncthreads();   // protect LDS tiles before next iteration overwrites
  }

  const float inv = 1.0f / lrun;
#pragma unroll
  for (int ct = 0; ct < 4; ++ct)
#pragma unroll
    for (int r = 0; r < 8; ++r)
      Ao[((size_t)bh * C_ + ct * 16 + r + 8 * g) * L_ + qbase + n] = o[ct][r] * inv;
}

// ---------------------------------------------------------------------------
// Depthwise k=3 (pad 1) and k=15 (pad 7) over l, per channel, on values (f32)
// ---------------------------------------------------------------------------
__global__ __launch_bounds__(256) void depthwise_kernel(
    const float* __restrict__ Vf, const float* __restrict__ dw0,
    const float* __restrict__ db0, const float* __restrict__ dw1,
    const float* __restrict__ db1, float* __restrict__ t0,
    float* __restrict__ t1) {
  int idx = blockIdx.x * blockDim.x + threadIdx.x;   // BH_*C_*L_ = 4194304
  int l  = idx & (L_ - 1);
  int c  = (idx >> 11) & (C_ - 1);
  const float* vr = Vf + (size_t)(idx >> 11) * L_;   // (bh*64 + c) row

  float a0 = db0[c];
#pragma unroll
  for (int j = 0; j < 3; ++j) {
    int p = l + j - 1;
    if (p >= 0 && p < L_) a0 = fmaf(dw0[c * 3 + j], vr[p], a0);
  }
  float a1 = db1[c];
#pragma unroll
  for (int j = 0; j < 15; ++j) {
    int p = l + j - 7;
    if (p >= 0 && p < L_) a1 = fmaf(dw1[c * 15 + j], vr[p], a1);
  }
  t0[idx] = a0;
  t1[idx] = a1;
}

// ---------------------------------------------------------------------------
// Pointwise 64x64 for both branches + gate softmax combine -> dyn (bh, c, l)
// ---------------------------------------------------------------------------
__global__ __launch_bounds__(256) void pw_dyn_kernel(
    const float* __restrict__ t0, const float* __restrict__ t1,
    const float* __restrict__ pw0, const float* __restrict__ pb0,
    const float* __restrict__ pw1, const float* __restrict__ pb1,
    const float* __restrict__ gate, float* __restrict__ dyn) {
  int idx = blockIdx.x * blockDim.x + threadIdx.x;   // BH_*C_*L_
  int l  = idx & (L_ - 1);
  int oc = (idx >> 11) & (C_ - 1);
  int bh = idx >> 17;

  float a0 = pb0[oc], a1 = pb1[oc];
  const float* w0 = pw0 + oc * C_;
  const float* w1 = pw1 + oc * C_;
  size_t base = (size_t)bh * C_ * L_ + l;
#pragma unroll 8
  for (int c = 0; c < C_; ++c) {
    a0 = fmaf(w0[c], t0[base + (size_t)c * L_], a0);
    a1 = fmaf(w1[c], t1[base + (size_t)c * L_], a1);
  }
  float e0 = __expf(gate[0]), e1 = __expf(gate[1]);
  float g0 = e0 / (e0 + e1);
  dyn[idx] = g0 * a0 + (1.0f - g0) * a1;
}

// ---------------------------------------------------------------------------
// unify_heads: out[b,o,l] = u_b[o] + sum_{ch<1024} u_w[o,ch] * cat[b,ch,l]
// cat = [attn (512 = head*64+c), dyn (512)], both stored (bh, c, l)
// ---------------------------------------------------------------------------
__global__ __launch_bounds__(256) void unify_kernel(
    const float* __restrict__ attn, const float* __restrict__ dyn,
    const float* __restrict__ uw, const float* __restrict__ ub,
    float* __restrict__ out) {
  int idx = blockIdx.x * blockDim.x + threadIdx.x;   // B_*C_*L_ = 524288
  int l = idx & (L_ - 1);
  int o = (idx >> 11) & (C_ - 1);
  int b = idx >> 17;

  float acc = ub[o];
  const float* wr = uw + o * (2 * OC_);
  size_t base = (size_t)b * H_ * C_ * L_ + l;        // (b*8 heads) * 64 * L
#pragma unroll 4
  for (int ch = 0; ch < OC_; ++ch) {
    size_t off = base + (size_t)ch * L_;             // ch = head*64 + c contiguous rows
    acc = fmaf(wr[ch], attn[off], acc);
    acc = fmaf(wr[OC_ + ch], dyn[off], acc);
  }
  out[idx] = acc;
}

// ---------------------------------------------------------------------------
extern "C" void kernel_launch(void* const* d_in, const int* in_sizes, int n_in,
                              void* d_out, int out_size, void* d_ws, size_t ws_size,
                              hipStream_t stream) {
  const float* x      = (const float*)d_in[0];
  const float* q_dw   = (const float*)d_in[1];
  const float* q_db   = (const float*)d_in[2];
  const float* q_pw   = (const float*)d_in[3];
  const float* q_pb   = (const float*)d_in[4];
  const float* k_dw   = (const float*)d_in[5];
  const float* k_db   = (const float*)d_in[6];
  const float* k_pw   = (const float*)d_in[7];
  const float* k_pb   = (const float*)d_in[8];
  const float* v_dw   = (const float*)d_in[9];
  const float* v_db   = (const float*)d_in[10];
  const float* v_pw   = (const float*)d_in[11];
  const float* v_pb   = (const float*)d_in[12];
  const float* dc0_dw = (const float*)d_in[13];
  const float* dc0_db = (const float*)d_in[14];
  const float* dc0_pw = (const float*)d_in[15];
  const float* dc0_pb = (const float*)d_in[16];
  const float* dc1_dw = (const float*)d_in[17];
  const float* dc1_db = (const float*)d_in[18];
  const float* dc1_pw = (const float*)d_in[19];
  const float* dc1_pb = (const float*)d_in[20];
  const float* gate   = (const float*)d_in[21];
  const float* u_w    = (const float*)d_in[22];
  const float* u_b    = (const float*)d_in[23];
  float* out = (float*)d_out;

  // Workspace carve-up (bytes)
  const size_t NE   = (size_t)BH_ * C_ * L_;           // 4194304 elements
  char* ws = (char*)d_ws;
  _Float16* Qt   = (_Float16*)(ws);                    // 8 MB  (bh, l, c) f16
  _Float16* Kt   = (_Float16*)(ws + NE * 2);           // 8 MB  (bh, l, c) f16
  _Float16* Vh   = (_Float16*)(ws + NE * 4);           // 8 MB  (bh, c, l) f16
  float*    Vf   = (float*)   (ws + NE * 6);           // 16 MB (bh, c, l) f32
  float*    t0   = (float*)   (ws + NE * 10);          // 16 MB
  float*    t1   = (float*)   (ws + NE * 14);          // 16 MB
  float*    attn = (float*)   (ws + NE * 18);          // 16 MB (bh, c, l) f32
  float*    dyn  = (float*)   (ws + NE * 22);          // 16 MB

  const int threads = 256;
  const int projBlocks = (B_ * OC_ * L_) / threads;    // 16384

  qkv_proj_kernel<<<projBlocks, threads, 0, stream>>>(x, q_dw, q_db, q_pw, q_pb, Qt, nullptr, 0);
  qkv_proj_kernel<<<projBlocks, threads, 0, stream>>>(x, k_dw, k_db, k_pw, k_pb, Kt, nullptr, 1);
  qkv_proj_kernel<<<projBlocks, threads, 0, stream>>>(x, v_dw, v_db, v_pw, v_pb, Vh, Vf, 2);

  depthwise_kernel<<<(int)(NE / threads), threads, 0, stream>>>(Vf, dc0_dw, dc0_db, dc1_dw, dc1_db, t0, t1);

  dim3 agrid(L_ / 16 / 8, BH_);                        // (16, 32), 8 waves/block
  flash_attn_kernel<<<agrid, threads, 0, stream>>>(Qt, Kt, Vh, attn);

  pw_dyn_kernel<<<(int)(NE / threads), threads, 0, stream>>>(t0, t1, dc0_pw, dc0_pb, dc1_pw, dc1_pb, gate, dyn);

  unify_kernel<<<(B_ * C_ * L_) / threads, threads, 0, stream>>>(attn, dyn, u_w, u_b, out);
}
